// CGMModelDiscriminator_61692910240320
// MI455X (gfx1250) — matrix operand order repaired and checked by reading, not compiled, not run on hardware
//
#include <hip/hip_runtime.h>
#include <hip/hip_bf16.h>

typedef _Float16 half_t;
typedef __attribute__((ext_vector_type(16))) _Float16 v16h;
typedef __attribute__((ext_vector_type(8)))  _Float16 v8h;
typedef __attribute__((ext_vector_type(8)))  float    v8f;

typedef __attribute__((ext_vector_type(4))) unsigned tdmv4u;
typedef __attribute__((ext_vector_type(8))) int      tdmv8i;
typedef __attribute__((ext_vector_type(4))) int      tdmv4i;

union frag16 { v16h v; v8h h[2]; };

#define LOG2PI_F 1.8378770664093453f

#if __has_builtin(__builtin_amdgcn_tensor_load_to_lds) && __has_builtin(__builtin_amdgcn_s_wait_tensorcnt)
#define USE_TDM 1
#else
#define USE_TDM 0
#endif

// LDS B-strip layout: 64 cols x 264 halves (256 K + 4-dword pad per row).
// 528B col stride = 132 dwords -> bank step 4 -> ds_load_b128 conflict-free.
#define BCOL_STRIDE 264

// ---------------------------------------------------------------------------
// Prep: transpose weights to N-major f16 (WMMA B-fragment friendly),
// convert latents to f16, fold +0.5 into bp.
// ---------------------------------------------------------------------------
__global__ void k_prep_weights(const float* __restrict__ W0, const float* __restrict__ W1,
                               const float* __restrict__ W2, const float* __restrict__ W3,
                               const float* __restrict__ W4, const float* __restrict__ Wp,
                               half_t* __restrict__ wt, half_t* __restrict__ wpt) {
    int idx = blockIdx.x * blockDim.x + threadIdx.x;
    if (idx < 5 * 65536) {
        int l = idx >> 16, r = idx & 65535;
        int n = r >> 8, k = r & 255;
        const float* W = (l == 0) ? W0 : (l == 1) ? W1 : (l == 2) ? W2 : (l == 3) ? W3 : W4;
        wt[idx] = (half_t)W[k * 256 + n];              // wt[l][n][k] = W_l[k][n]
    } else if (idx < 6 * 65536) {
        int r = idx - 5 * 65536;
        int c = r >> 8, h = r & 255;                   // c = n*2+k
        wpt[r] = (half_t)Wp[(c >> 1) * 512 + h * 2 + (c & 1)];  // wpt[c][h] = Wp[n][h][k]
    }
}

__global__ void k_prep_misc(const float* __restrict__ latents, const float* __restrict__ bp,
                            half_t* __restrict__ actA, float* __restrict__ bp2) {
    int idx = blockIdx.x * blockDim.x + threadIdx.x;
    if (idx < 262144) {
        actA[idx] = (half_t)latents[idx];
    } else if (idx < 262144 + 256) {
        int c = idx - 262144;
        bp2[c] = bp[c] + 0.5f;                         // fold the +0.5 of mu into bias
    }
}

// ---------------------------------------------------------------------------
// WMMA GEMM: C(1024x256) = act( A(1024x256,f16) @ Bt^T + bias )
//   A  : row-major f16, stride 256 (direct global b128 fragment loads)
//   Bt : N-major f16 (Bt[n*256+k] == W[k][n])
// The 64-col B strip shared by all 8 waves of the block is DMA'd to LDS once
// via tensor_load_to_lds (TDM), with pad_interval=128dw/pad_amount=4dw giving
// the conflict-free 264-half column stride. Fragments then come from LDS.
// ---------------------------------------------------------------------------
__global__ __launch_bounds__(256) void k_gemm_wmma(
    const half_t* __restrict__ A, const half_t* __restrict__ Bt,
    const float* __restrict__ bias,
    half_t* __restrict__ outH, float* __restrict__ outF, int relu) {
    __shared__ half_t bS[64 * BCOL_STRIDE];

    const int lane = threadIdx.x & 31;
    const int wave = threadIdx.x >> 5;
    const int l15  = lane & 15;
    const int g    = lane >> 4;
    const int mBase  = (blockIdx.x * 8 + wave) * 16;
    const int nGroup = blockIdx.y * 64;

    const half_t* bSrc = Bt + nGroup * 256;   // 64 rows x 256 halves, contiguous

#if USE_TDM
    if (threadIdx.x < 32) {                   // wave 0 issues the DMA (EXEC ignored by TDM)
        const unsigned lds = (unsigned)(unsigned long long)(&bS[0]);  // LDS aperture: low 32 bits
        const unsigned long long ga = (unsigned long long)bSrc;
        tdmv4u g0;
        g0[0] = 1u;                                              // count=1, user mode
        g0[1] = lds;                                             // lds_addr
        g0[2] = (unsigned)(ga & 0xffffffffu);                    // global_addr[31:0]
        g0[3] = (unsigned)((ga >> 32) & 0x01ffffffu) | 0x80000000u; // addr[56:32] | type=2
        tdmv8i g1;
        g1[0] = (1 << 16)      // data_size = 2 bytes
              | (1 << 20)      // pad_enable
              | (6 << 22)      // pad_interval: 128 dwords
              | (3 << 25);     // pad_amount: 4 dwords
        g1[1] = (256 << 16);   // tensor_dim0 = 256
        g1[2] = (64  << 16);   // tensor_dim1 = 64
        g1[3] = (256 << 16);   // tile_dim0   = 256
        g1[4] = 64;            // tile_dim1   = 64, tile_dim2 = 0
        g1[5] = 256;           // tensor_dim0_stride = 256
        g1[6] = 0;
        g1[7] = 0;
        tdmv4i gz = {0, 0, 0, 0};
#if defined(__clang_major__) && (__clang_major__ >= 23)
        tdmv8i gz8 = {0, 0, 0, 0, 0, 0, 0, 0};
        __builtin_amdgcn_tensor_load_to_lds(g0, g1, gz, gz, gz8, 0);
#else
        __builtin_amdgcn_tensor_load_to_lds(g0, g1, gz, gz, 0);
#endif
        __builtin_amdgcn_s_wait_tensorcnt((short)0);
    }
#else
    // Fallback: cooperative copy (global_load_b128 + ds_store_b128)
    for (int i = threadIdx.x; i < 2048; i += 256) {
        int c = i >> 5, ck = i & 31;
        *(v8h*)&bS[c * BCOL_STRIDE + ck * 8] = *(const v8h*)(bSrc + c * 256 + ck * 8);
    }
#endif
    __syncthreads();

    // A-fragment addressing per ISA 16-bit A 16x32 layout:
    //   row = mBase + (lane&15); halves 0..7 -> K = k0 + 8*g .. +7 ; halves 8..15 -> +16
    const half_t* aPtr = A + (mBase + l15) * 256 + g * 8;

    v8f acc[4];
    #pragma unroll
    for (int j = 0; j < 4; j++) { v8f z = {}; acc[j] = z; }

    #pragma unroll
    for (int k0 = 0; k0 < 256; k0 += 32) {
        frag16 a;
        a.h[0] = *(const v8h*)(aPtr + k0);
        a.h[1] = *(const v8h*)(aPtr + k0 + 16);
        #pragma unroll
        for (int j = 0; j < 4; j++) {
            // B-fragment from LDS: col = lane&15, halves 0..7 -> K = k0 + 8*g .. +7 ; halves 8..15 -> +16
            const half_t* bPtr = &bS[(j * 16 + l15) * BCOL_STRIDE + k0 + g * 8];
            frag16 bb;
            bb.h[0] = *(const v8h*)(bPtr);
            bb.h[1] = *(const v8h*)(bPtr + 16);
            acc[j] = __builtin_amdgcn_wmma_f32_16x16x32_f16(
                false, a.v, false, bb.v, (short)0, acc[j], false, false);
        }
    }

    // C/D layout: VGPR v -> M = mBase + v + 8*g, N = nGroup + j*16 + (lane&15)
    #pragma unroll
    for (int j = 0; j < 4; j++) {
        const int col = nGroup + j * 16 + l15;
        const float bv = bias[col];
        #pragma unroll
        for (int v = 0; v < 8; v++) {
            const int row = mBase + v + g * 8;
            float val = acc[j][v] + bv;
            if (relu) val = fmaxf(val, 0.f);
            if (outH) outH[row * 256 + col] = (half_t)val;
            else      outF[row * 256 + col] = val;
        }
    }
}

// ---------------------------------------------------------------------------
// Loss kernel: one block per batch b; 256 threads = (n in [0,128), s in {0,1}).
// Computes sampled, realLoss, lossA, genLoss, lossB.
// ---------------------------------------------------------------------------
__global__ __launch_bounds__(256) void k_loss(
    const float* __restrict__ mu,       // (B, 256) columns = n*2+k (already +0.5+bp)
    const float* __restrict__ eps,      // (2, 131072, 2)
    const float* __restrict__ realPts,  // (B, 2)
    const float* __restrict__ predMu,   // (B, 2)
    const float* __restrict__ predSc,   // (B, 2)
    float* __restrict__ o_sampled, float* __restrict__ o_gen,
    float* __restrict__ o_real, float* __restrict__ o_lossA,
    float* __restrict__ o_lossB) {
    __shared__ float muS[256];
    __shared__ float invDevS[128];
    __shared__ float cS[128];
    __shared__ float realLossS[128];
    __shared__ float laS[2][128];
    __shared__ float glS[2][128];

    const int b = blockIdx.x;
    const int t = threadIdx.x;

    muS[t] = mu[b * 256 + t];
    if (t < 128) {
        float dev = 0.001f + 0.009f * (float)t / 127.f;   // linspace(1e-3, 1e-2, 128)
        invDevS[t] = 1.f / dev;
        cS[t] = -2.f * __logf(dev) - LOG2PI_F;
    }
    __syncthreads();

    const int n = t & 127;
    const int s = t >> 7;
    const float dev_n = 0.001f + 0.009f * (float)n / 127.f;
    const float mu0 = muS[2 * n], mu1 = muS[2 * n + 1];

    // sampled[(s*N+n), b, :] = mu + dev*eps[s, b*N+n, :]
    const long eIdx = ((long)s * 131072 + (long)b * 128 + n) * 2;
    const float x0 = mu0 + dev_n * eps[eIdx];
    const float x1 = mu1 + dev_n * eps[eIdx + 1];
    const long sIdx = (((long)(s * 128 + n)) * 1024 + b) * 2;
    o_sampled[sIdx]     = x0;
    o_sampled[sIdx + 1] = x1;

    // realLoss (per (b,n), compute once with s==0 threads)
    if (s == 0) {
        float z0 = (realPts[b * 2]     - mu0) / dev_n;
        float z1 = (realPts[b * 2 + 1] - mu1) / dev_n;
        float lp = -0.5f * (z0 * z0 + z1 * z1) - 2.f * __logf(dev_n) - LOG2PI_F;
        float rl = 1.f / (1.f + __expf(-lp));
        realLossS[n] = rl;
        o_real[b * 128 + n] = rl;
    }

    // lossA / genLoss pieces: logprob under predicted dist
    const float pm0 = predMu[b * 2], pm1 = predMu[b * 2 + 1];
    const float ps0 = predSc[b * 2], ps1 = predSc[b * 2 + 1];
    float zp0 = (x0 - pm0) / ps0, zp1 = (x1 - pm1) / ps1;
    float la = -0.5f * (zp0 * zp0 + zp1 * zp1) - __logf(ps0) - __logf(ps1) - LOG2PI_F;
    float siga = 1.f / (1.f + __expf(la));          // sigmoid(-la)
    float dA = log1pf(0.5f) - log1pf(siga);
    laS[s][n] = dA * dA;
    glS[s][n] = 1.f / (1.f + __expf(-la));          // sigmoid(la)

    // lossB: softmax diagonal over all N gaussians (single pass; lb <= ~12 so
    // exp never overflows f32; underflow to 0 is harmless)
    float sumexp = 0.f, en = 0.f;
    #pragma unroll 4
    for (int n2 = 0; n2 < 128; n2++) {
        float z0 = (x0 - muS[2 * n2])     * invDevS[n2];
        float z1 = (x1 - muS[2 * n2 + 1]) * invDevS[n2];
        float lb = cS[n2] - 0.5f * (z0 * z0 + z1 * z1);
        float e = __expf(lb);
        sumexp += e;
        if (n2 == n) en = e;
    }
    float dB = 1.f - en / sumexp;
    o_lossB[((long)(b * 128 + n)) * 2 + s] = dB * dB;

    __syncthreads();
    if (s == 0) {
        o_lossA[b * 128 + n] = 0.5f * (laS[0][n] + laS[1][n]);
        float gmean = 0.5f * (glS[0][n] + glS[1][n]);
        o_gen[b * 128 + n] = gmean * (1.f - realLossS[n]);
    }
}

// ---------------------------------------------------------------------------
extern "C" void kernel_launch(void* const* d_in, const int* in_sizes, int n_in,
                              void* d_out, int out_size, void* d_ws, size_t ws_size,
                              hipStream_t stream) {
    const float* latents = (const float*)d_in[0];
    const float* realPts = (const float*)d_in[1];
    const float* predMu  = (const float*)d_in[2];
    const float* predSc  = (const float*)d_in[3];
    const float* eps     = (const float*)d_in[4];
    const float* W[5]    = {(const float*)d_in[5],  (const float*)d_in[7],
                            (const float*)d_in[9],  (const float*)d_in[11],
                            (const float*)d_in[13]};
    const float* bvec[5] = {(const float*)d_in[6],  (const float*)d_in[8],
                            (const float*)d_in[10], (const float*)d_in[12],
                            (const float*)d_in[14]};
    const float* Wp = (const float*)d_in[15];
    const float* bp = (const float*)d_in[16];

    // workspace layout (all 16B aligned): ~2.9 MB total
    half_t* wt   = (half_t*)d_ws;            // 5*65536 f16  (transposed W0..W4)
    half_t* wpt  = wt + 5 * 65536;           // 65536 f16    (transposed Wp)
    float*  bp2  = (float*)(wpt + 65536);    // 256 f32      (bp + 0.5)
    half_t* actA = (half_t*)(bp2 + 256);     // 262144 f16   activations ping
    half_t* actB = actA + 262144;            // 262144 f16   activations pong
    float*  muW  = (float*)(actB + 262144);  // 262144 f32   mu matrix

    k_prep_weights<<<6 * 65536 / 256, 256, 0, stream>>>(W[0], W[1], W[2], W[3], W[4], Wp, wt, wpt);
    k_prep_misc<<<(262144 + 256 + 255) / 256, 256, 0, stream>>>(latents, bp, actA, bp2);

    dim3 ggrid(8, 4);  // 64 M-tiles / 8 waves, 4 N-groups of 64
    half_t* src = actA;
    half_t* dst = actB;
    for (int l = 0; l < 5; l++) {
        k_gemm_wmma<<<ggrid, 256, 0, stream>>>(src, wt + l * 65536, bvec[l], dst, nullptr, 1);
        half_t* tmp = src; src = dst; dst = tmp;
    }
    // src now holds h (f16); projection -> mu (f32), no relu
    k_gemm_wmma<<<ggrid, 256, 0, stream>>>(src, wpt, bp2, nullptr, muW, 0);

    float* out = (float*)d_out;
    k_loss<<<1024, 256, 0, stream>>>(muW, eps, realPts, predMu, predSc,
                                     out,                            // sampled (524288)
                                     out + 524288,                   // genLoss (131072)
                                     out + 524288 + 131072,          // realLoss (131072)
                                     out + 524288 + 2 * 131072,      // lossA   (131072)
                                     out + 524288 + 3 * 131072);     // lossB   (262144)
}
